// SlateEvaluator_36447092474580
// MI455X (gfx1250) — matrix operand order repaired and checked by reading, not compiled, not run on hardware
//
#include <hip/hip_runtime.h>

// ---------------------------------------------------------------------------
// Types for CDNA5 WMMA (wave32).  D = A(16x32 bf16) x B(32x16 bf16) + C(f32)
// ---------------------------------------------------------------------------
typedef __attribute__((ext_vector_type(16))) __bf16 v16bf;
typedef __attribute__((ext_vector_type(8)))  __bf16 v8bf;
typedef __attribute__((ext_vector_type(8)))  float  v8f;

#define Bdim 4096
#define Kdim 32
#define STATE_DIM 512
#define ENC_DIM 128
#define Hdim 128
#define MLP_IN 1024
#define H1 1024
#define H2 512
#define ROWS (Bdim * Kdim)          // 131072
#define EPSV 1e-5f

__device__ __forceinline__ v8f wmma_bf16(v16bf a, v16bf b, v8f c) {
  // emits v_wmma_f32_16x16x32_bf16
  return __builtin_amdgcn_wmma_f32_16x16x32_bf16(false, a, false, b, (short)0, c,
                                                 false, false);
}

// A-matrix fragment (16x32, row-major in LDS, leading dim = ld halves).
// ISA 7.12.2: lanes 0-15 hold rows M=0..15 K-chunks {0..7,16..23},
//             lanes 16-31 hold K-chunks {8..15,24..31}.
__device__ __forceinline__ v16bf load_a_frag(const __bf16* p, int ld) {
  int l   = threadIdx.x & 31;
  int row = l & 15;
  int hb  = (l >> 4) * 8;
  union { v16bf v; v8bf h[2]; } u;
  u.h[0] = *(const v8bf*)(p + row * ld + hb);
  u.h[1] = *(const v8bf*)(p + row * ld + 16 + hb);
  return u.v;
}

// B-matrix fragment for D = X @ W^T : B[k][n] = W[n][k], W row-major (N x K)
// ISA 7.12.2: lane n = l%16 holds k {0..7,16..23} (l<16) or {8..15,24..31}.
__device__ __forceinline__ v16bf load_b_frag(const __bf16* p, int ld) {
  int l   = threadIdx.x & 31;
  int col = l & 15;
  int hb  = (l >> 4) * 8;
  union { v16bf v; v8bf h[2]; } u;
  u.h[0] = *(const v8bf*)(p + col * ld + hb);
  u.h[1] = *(const v8bf*)(p + col * ld + 16 + hb);
  return u.v;
}

__device__ __forceinline__ float sigmoidf_(float x) {
  return 1.0f / (1.0f + __expf(-x));
}
__device__ __forceinline__ float tanh_fast(float x) {
  return 1.0f - 2.0f / (__expf(2.0f * x) + 1.0f);
}

// ---------------------------------------------------------------------------
// 0) zero stats + reg slot
// ---------------------------------------------------------------------------
__global__ void zero_kernel(float* stats, int n, float* reg_out) {
  int i = blockIdx.x * 256 + threadIdx.x;
  if (i < n) stats[i] = 0.0f;
  if (i == 0) *reg_out = 0.0f;
}

// ---------------------------------------------------------------------------
// 1) regularizer: atomicAdd( mean(p*p) )
// ---------------------------------------------------------------------------
__global__ void reg_kernel(const float* __restrict__ p, int n, float inv_n,
                           float* out) {
  __shared__ float red[256];
  float s = 0.0f;
  for (int i = blockIdx.x * 256 + threadIdx.x; i < n; i += gridDim.x * 256) {
    float v = p[i];
    s += v * v;
  }
  red[threadIdx.x] = s;
  __syncthreads();
  for (int o = 128; o > 0; o >>= 1) {
    if (threadIdx.x < o) red[threadIdx.x] += red[threadIdx.x + o];
    __syncthreads();
  }
  if (threadIdx.x == 0) atomicAdd(out, red[0] * inv_n);
}

// ---------------------------------------------------------------------------
// 2) f32 -> bf16 weight convert
// ---------------------------------------------------------------------------
__global__ void cvt_bf16_kernel(const float* __restrict__ src,
                                __bf16* __restrict__ dst, int n) {
  int i = blockIdx.x * 256 + threadIdx.x;
  if (i < n) dst[i] = (__bf16)src[i];
}

// ---------------------------------------------------------------------------
// 3) attention + assemble X = [S | Of | Or | V | A]  (O filled by LSTM later)
//    one block per batch element b
// ---------------------------------------------------------------------------
__global__ void attn_build_kernel(const float* __restrict__ state,
                                  const float* __restrict__ action,
                                  __bf16* __restrict__ X) {
  __shared__ float Vf[32 * 128];
  __shared__ float sc[32 * 32];
  int b = blockIdx.x, tid = threadIdx.x;

  for (int i = tid; i < 32 * 128; i += 256) {
    int k = i >> 7, d = i & 127;
    float v = action[((size_t)b * 32 + k) * 128 + d];
    Vf[i] = v;
    X[((size_t)b * 32 + k) * 1024 + 768 + d] = (__bf16)v;
  }
  for (int i = tid; i < 32 * 512; i += 256) {
    int k = i >> 9, c = i & 511;
    X[((size_t)b * 32 + k) * 1024 + c] = (__bf16)state[(size_t)b * 512 + c];
  }
  __syncthreads();
  for (int i = tid; i < 32 * 32; i += 256) {
    int k = i >> 5, l = i & 31;
    float s = 0.0f;
    for (int d = 0; d < 128; ++d) s += Vf[k * 128 + d] * Vf[l * 128 + d];
    sc[i] = s;
  }
  __syncthreads();
  if (tid < 32) {
    float mx = -1e30f;
    for (int l = 0; l < 32; ++l) mx = fmaxf(mx, sc[tid * 32 + l]);
    float sum = 0.0f;
    for (int l = 0; l < 32; ++l) {
      float e = __expf(sc[tid * 32 + l] - mx);
      sc[tid * 32 + l] = e;
      sum += e;
    }
    float inv = 1.0f / sum;
    for (int l = 0; l < 32; ++l) sc[tid * 32 + l] *= inv;
  }
  __syncthreads();
  for (int i = tid; i < 32 * 128; i += 256) {
    int k = i >> 7, d = i & 127;
    float a = 0.0f;
    for (int l = 0; l < 32; ++l) a += sc[k * 32 + l] * Vf[l * 128 + d];
    X[((size_t)b * 32 + k) * 1024 + 896 + d] = (__bf16)a;
  }
}

// ---------------------------------------------------------------------------
// 4) LSTM, one direction.  Block = 16 batch rows, 256 threads (8 waves).
//    [w_ih|w_hh] resident in LDS with padded leading dim (bank-conflict-free
//    B-fragment reads on the 64-bank LDS).  Cell state c lives in registers
//    (per-thread (row,col) ownership is step-invariant).
//    LDS: Wc 512x264 halves (270336B) + bias 2048B + gates 32768B
//         + At 16x264 halves (8448B)  = 313600B  (< 320KB/WGP)
// ---------------------------------------------------------------------------
#define WLD 264
#define ALD 264
__global__ void lstm_kernel(const float* __restrict__ action,
                            const float* __restrict__ w_ih,
                            const float* __restrict__ w_hh,
                            const float* __restrict__ b_ih,
                            const float* __restrict__ b_hh,
                            __bf16* __restrict__ X, int dir) {
  extern __shared__ char smem[];
  __bf16* Wc   = (__bf16*)smem;                         // 512 x WLD halves
  float*  bias = (float*)(smem + 512 * WLD * 2);        // 512 f32
  float*  gat  = bias + 512;                            // 16*512 f32
  __bf16* At   = (__bf16*)(gat + 16 * 512);             // 16 x ALD halves

  int tid = threadIdx.x, lane = tid & 31, wave = tid >> 5;
  int b0 = blockIdx.x * 16;
  int colBase = wave * 64;  // this wave's 4 gate-column tiles

  for (int i = tid; i < 512 * 256; i += 256) {
    int r = i >> 8, c = i & 255;
    float v = (c < 128) ? w_ih[r * 128 + c] : w_hh[r * 128 + (c - 128)];
    Wc[r * WLD + c] = (__bf16)v;
  }
  for (int i = tid; i < 512; i += 256) bias[i] = b_ih[i] + b_hh[i];
  for (int i = tid; i < 16 * 128; i += 256)
    At[(i >> 7) * ALD + 128 + (i & 127)] = (__bf16)0.0f;  // h0 = 0

  float creg[8];  // cell state: element e <-> flat index tid + 256*e
#pragma unroll
  for (int e = 0; e < 8; ++e) creg[e] = 0.0f;

  int outOff = dir ? 640 : 512;
  for (int step = 0; step < 32; ++step) {
    int ks = dir ? (31 - step) : step;
    for (int i = tid; i < 16 * 128; i += 256) {
      int r = i >> 7, c = i & 127;
      At[r * ALD + c] = (__bf16)action[((size_t)(b0 + r) * 32 + ks) * 128 + c];
    }
    __syncthreads();

    v8f acc[4];
#pragma unroll
    for (int j = 0; j < 4; ++j) {
      float bv = bias[colBase + j * 16 + (lane & 15)];
#pragma unroll
      for (int v = 0; v < 8; ++v) acc[j][v] = bv;
    }
#pragma unroll
    for (int kk = 0; kk < 8; ++kk) {
      // batch all LDS fragment loads, then issue 4 back-to-back WMMAs
      v16bf a = load_a_frag(At + kk * 32, ALD);
      v16bf bf[4];
#pragma unroll
      for (int j = 0; j < 4; ++j)
        bf[j] = load_b_frag(Wc + (colBase + j * 16) * WLD + kk * 32, WLD);
#pragma unroll
      for (int j = 0; j < 4; ++j) acc[j] = wmma_bf16(a, bf[j], acc[j]);
    }
#pragma unroll
    for (int j = 0; j < 4; ++j)
#pragma unroll
      for (int v = 0; v < 8; ++v)
        gat[((lane >> 4) * 8 + v) * 512 + colBase + j * 16 + (lane & 15)] =
            acc[j][v];
    __syncthreads();

#pragma unroll
    for (int e = 0; e < 8; ++e) {
      int i = tid + 256 * e;
      int r = i >> 7, j = i & 127;
      float gi = gat[r * 512 + j];
      float gf = gat[r * 512 + 128 + j];
      float gg = gat[r * 512 + 256 + j];
      float go = gat[r * 512 + 384 + j];
      float c = sigmoidf_(gf) * creg[e] + sigmoidf_(gi) * tanh_fast(gg);
      creg[e] = c;
      float h = sigmoidf_(go) * tanh_fast(c);
      At[r * ALD + 128 + j] = (__bf16)h;
      X[((size_t)(b0 + r) * 32 + ks) * 1024 + outOff + j] = (__bf16)h;
    }
    __syncthreads();
  }
}

// ---------------------------------------------------------------------------
// 5) bf16 WMMA GEMM:  Y(M x N) = act(A) @ Bw^T + bias, act = optional BN+ReLU
//    block = 256 threads (8 waves), 128x128 output tile, K-chunks of 32,
//    double-buffered LDS (1 barrier per K-step) + global_prefetch of k+2.
//    Grid: x = N-tiles (fast), y = M-tiles -> A row-panel stays L2-resident
//    and the activation matrix streams from HBM exactly once.
// ---------------------------------------------------------------------------
#define GLD 40  // LDS leading dim (halves): 80B rows, conflict-free frag reads
__global__ void gemm_bn_kernel(const __bf16* __restrict__ A,
                               const __bf16* __restrict__ Bw,
                               const float* __restrict__ bias,
                               const float* __restrict__ scale,
                               const float* __restrict__ shift,
                               __bf16* __restrict__ Y, int Kt, int N,
                               int use_bn) {
  __shared__ __align__(16) __bf16 As[2][128 * GLD];
  __shared__ __align__(16) __bf16 Bs[2][128 * GLD];
  __shared__ float scl[1024];
  __shared__ float shf[1024];

  int tid = threadIdx.x, lane = tid & 31, wave = tid >> 5;
  int n0 = blockIdx.x * 128, m0 = blockIdx.y * 128;
  int wy = wave >> 1, wx = wave & 1;  // 4x2 wave grid over (M,N)

  if (use_bn)
    for (int i = tid; i < Kt; i += 256) {
      scl[i] = scale[i];
      shf[i] = shift[i];
    }

  v8f acc[2][4];
#pragma unroll
  for (int mi = 0; mi < 2; ++mi)
#pragma unroll
    for (int j = 0; j < 4; ++j)
#pragma unroll
      for (int v = 0; v < 8; ++v) acc[mi][j][v] = 0.0f;

  int r = tid >> 1, cc = (tid & 1) * 16;  // 32 halves per row, 16 per thread
  const __bf16* aRow = A + (size_t)(m0 + r) * Kt + cc;
  const __bf16* bRow = Bw + (size_t)(n0 + r) * Kt + cc;

  uint4 a0, a1, b0, b1;
  auto fetch = [&](int kk) {
    a0 = *(const uint4*)(aRow + kk);
    a1 = *(const uint4*)(aRow + kk + 8);
    b0 = *(const uint4*)(bRow + kk);
    b1 = *(const uint4*)(bRow + kk + 8);
    if (kk + 32 < Kt) {
      __builtin_prefetch(aRow + kk + 32, 0, 3);  // global_prefetch_b8
      __builtin_prefetch(bRow + kk + 32, 0, 3);
    }
  };
  auto store = [&](int kk, int buf) {
    if (!use_bn) {
      *(uint4*)&As[buf][r * GLD + cc]     = a0;
      *(uint4*)&As[buf][r * GLD + cc + 8] = a1;
    } else {
      union { uint4 u[2]; __bf16 h[16]; } ua;
      ua.u[0] = a0;
      ua.u[1] = a1;
#pragma unroll
      for (int i = 0; i < 16; ++i) {
        int col = kk + cc + i;
        float x = (float)ua.h[i] * scl[col] + shf[col];
        As[buf][r * GLD + cc + i] = (__bf16)fmaxf(x, 0.0f);
      }
    }
    *(uint4*)&Bs[buf][r * GLD + cc]     = b0;
    *(uint4*)&Bs[buf][r * GLD + cc + 8] = b1;
  };

  if (use_bn) __syncthreads();  // scl/shf visible before first store
  fetch(0);
  store(0, 0);
  __syncthreads();

  int cur = 0;
  for (int kk = 0; kk < Kt; kk += 32) {
    bool more = (kk + 32) < Kt;
    if (more) fetch(kk + 32);  // overlap next global tile with WMMAs

    v16bf bf[4];
#pragma unroll
    for (int j = 0; j < 4; ++j)
      bf[j] = load_b_frag(&Bs[cur][(wx * 64 + j * 16) * GLD], GLD);
#pragma unroll
    for (int mi = 0; mi < 2; ++mi) {
      v16bf af = load_a_frag(&As[cur][(wy * 32 + mi * 16) * GLD], GLD);
#pragma unroll
      for (int j = 0; j < 4; ++j) acc[mi][j] = wmma_bf16(af, bf[j], acc[mi][j]);
    }

    if (more) store(kk + 32, cur ^ 1);  // other buffer: no conflict w/ readers
    __syncthreads();
    cur ^= 1;
  }

#pragma unroll
  for (int mi = 0; mi < 2; ++mi)
#pragma unroll
    for (int j = 0; j < 4; ++j) {
      int n = n0 + wx * 64 + j * 16 + (lane & 15);
      float bv = bias[n];
#pragma unroll
      for (int v = 0; v < 8; ++v) {
        int m = m0 + wy * 32 + mi * 16 + (lane >> 4) * 8 + v;
        Y[(size_t)m * N + n] = (__bf16)(acc[mi][j][v] + bv);
      }
    }
}

// ---------------------------------------------------------------------------
// 6) column stats (sum, sumsq) for batchnorm over ROWS rows
// ---------------------------------------------------------------------------
__global__ void colstats_kernel(const __bf16* __restrict__ Y, int N,
                                float* __restrict__ sum,
                                float* __restrict__ sumsq) {
  int tid = threadIdx.x;
  int per = N >> 8;  // 4 (N=1024) or 2 (N=512) cols per thread
  float s[4] = {0, 0, 0, 0}, q[4] = {0, 0, 0, 0};
  int r0 = blockIdx.x * 128;
  for (int rr = 0; rr < 128; ++rr) {
    const __bf16* row = Y + (size_t)(r0 + rr) * N + tid * per;
    for (int i = 0; i < per; ++i) {
      float x = (float)row[i];
      s[i] += x;
      q[i] += x * x;
    }
  }
  for (int i = 0; i < per; ++i) {
    atomicAdd(&sum[tid * per + i], s[i]);
    atomicAdd(&sumsq[tid * per + i], q[i]);
  }
}

__global__ void bn_params_kernel(const float* __restrict__ sum,
                                 const float* __restrict__ sumsq,
                                 const float* __restrict__ g,
                                 const float* __restrict__ be,
                                 float* __restrict__ scale,
                                 float* __restrict__ shift, float invM) {
  int t = threadIdx.x;
  float mu  = sum[t] * invM;
  float var = sumsq[t] * invM - mu * mu;
  float sc  = g[t] * rsqrtf(var + EPSV);
  scale[t]  = sc;
  shift[t]  = be[t] - mu * sc;
}

// ---------------------------------------------------------------------------
// 7) head: Q[row] = relu(bn2(Y2[row,:])) . W3 + b3   (one wave per row)
// ---------------------------------------------------------------------------
__global__ void head_kernel(const __bf16* __restrict__ Y2,
                            const float* __restrict__ scale,
                            const float* __restrict__ shift,
                            const float* __restrict__ W3,
                            const float* __restrict__ b3,
                            float* __restrict__ Q) {
  int lane = threadIdx.x & 31, wave = threadIdx.x >> 5;
  int row = blockIdx.x * 8 + wave;
  float p = 0.0f;
  for (int j = lane; j < H2; j += 32) {
    float x = (float)Y2[(size_t)row * H2 + j] * scale[j] + shift[j];
    p += fmaxf(x, 0.0f) * W3[j];
  }
#pragma unroll
  for (int o = 16; o > 0; o >>= 1) p += __shfl_xor(p, o, 32);
  if (lane == 0) Q[row] = p + b3[0];
}

// ---------------------------------------------------------------------------
// host launcher
// ---------------------------------------------------------------------------
extern "C" void kernel_launch(void* const* d_in, const int* in_sizes, int n_in,
                              void* d_out, int out_size, void* d_ws,
                              size_t ws_size, hipStream_t stream) {
  const float* state  = (const float*)d_in[0];
  const float* action = (const float*)d_in[1];
  const float* w_ih_f = (const float*)d_in[2];
  const float* w_hh_f = (const float*)d_in[3];
  const float* b_ih_f = (const float*)d_in[4];
  const float* b_hh_f = (const float*)d_in[5];
  const float* w_ih_r = (const float*)d_in[6];
  const float* w_hh_r = (const float*)d_in[7];
  const float* b_ih_r = (const float*)d_in[8];
  const float* b_hh_r = (const float*)d_in[9];
  const float* W1 = (const float*)d_in[10];
  const float* b1 = (const float*)d_in[11];
  const float* g1 = (const float*)d_in[12];
  const float* be1 = (const float*)d_in[13];
  const float* W2 = (const float*)d_in[14];
  const float* b2 = (const float*)d_in[15];
  const float* g2 = (const float*)d_in[16];
  const float* be2 = (const float*)d_in[17];
  const float* W3 = (const float*)d_in[18];
  const float* b3 = (const float*)d_in[19];

  float* Q_out   = (float*)d_out;           // ROWS floats
  float* reg_out = (float*)d_out + ROWS;    // 1 float

  // workspace carve (Y2 aliases X: X is dead after GEMM1)
  char* ws = (char*)d_ws;
  const size_t SX  = (size_t)ROWS * MLP_IN * 2;  // 256 MB bf16
  const size_t SY1 = (size_t)ROWS * H1 * 2;      // 256 MB
  __bf16* X   = (__bf16*)(ws);
  __bf16* Y1  = (__bf16*)(ws + SX);
  __bf16* Y2  = (__bf16*)(ws);                   // alias over X
  __bf16* W1b = (__bf16*)(ws + SX + SY1);
  __bf16* W2b = (__bf16*)(ws + SX + SY1 + (size_t)H1 * MLP_IN * 2);
  float* fbuf = (float*)(ws + SX + SY1 + (size_t)H1 * MLP_IN * 2 +
                         (size_t)H2 * H1 * 2);
  float* sum1 = fbuf;              // 1024
  float* sq1  = sum1 + 1024;       // 1024
  float* sum2 = sq1 + 1024;        // 512
  float* sq2  = sum2 + 512;        // 512  (stats block: 3072 f32, zeroed)
  float* scl1 = sq2 + 512;
  float* shf1 = scl1 + 1024;
  float* scl2 = shf1 + 1024;
  float* shf2 = scl2 + 512;

  // 0) zero accumulators
  zero_kernel<<<(3072 + 255) / 256, 256, 0, stream>>>(sum1, 3072, reg_out);

  // 1) regularizer over the 18 params (inputs 2..19)
  for (int i = 2; i < 20; ++i) {
    int n = in_sizes[i];
    int blocks = (n + 255) / 256;
    if (blocks > 512) blocks = 512;
    reg_kernel<<<blocks, 256, 0, stream>>>((const float*)d_in[i], n,
                                           1.0f / (float)n, reg_out);
  }

  // 2) weight conversion to bf16
  cvt_bf16_kernel<<<(H1 * MLP_IN + 255) / 256, 256, 0, stream>>>(W1, W1b,
                                                                 H1 * MLP_IN);
  cvt_bf16_kernel<<<(H2 * H1 + 255) / 256, 256, 0, stream>>>(W2, W2b, H2 * H1);

  // 3) attention + X assembly (S, V, A slices)
  attn_build_kernel<<<Bdim, 256, 0, stream>>>(state, action, X);

  // 4) bidirectional LSTM (fills Of, Or slices of X) — 313600B dynamic LDS
  const size_t lstm_lds = (size_t)512 * WLD * 2 + 512 * 4 + 16 * 512 * 4 +
                          (size_t)16 * ALD * 2;
  lstm_kernel<<<Bdim / 16, 256, lstm_lds, stream>>>(action, w_ih_f, w_hh_f,
                                                    b_ih_f, b_hh_f, X, 0);
  lstm_kernel<<<Bdim / 16, 256, lstm_lds, stream>>>(action, w_ih_r, w_hh_r,
                                                    b_ih_r, b_hh_r, X, 1);

  // 5) layer 1: Y1 = X @ W1^T + b1   (grid.x = N-tiles, grid.y = M-tiles)
  dim3 g1d(H1 / 128, ROWS / 128);
  gemm_bn_kernel<<<g1d, 256, 0, stream>>>(X, W1b, b1, nullptr, nullptr, Y1,
                                          MLP_IN, H1, 0);
  colstats_kernel<<<ROWS / 128, 256, 0, stream>>>(Y1, H1, sum1, sq1);
  bn_params_kernel<<<1, H1, 0, stream>>>(sum1, sq1, g1, be1, scl1, shf1,
                                         1.0f / (float)ROWS);

  // 6) layer 2: Y2 = relu(bn1(Y1)) @ W2^T + b2   (BN+ReLU folded into A-load)
  dim3 g2d(H2 / 128, ROWS / 128);
  gemm_bn_kernel<<<g2d, 256, 0, stream>>>(Y1, W2b, b2, scl1, shf1, Y2, H1, H2,
                                          1);
  colstats_kernel<<<ROWS / 128, 256, 0, stream>>>(Y2, H2, sum2, sq2);
  bn_params_kernel<<<1, H2, 0, stream>>>(sum2, sq2, g2, be2, scl2, shf2,
                                         1.0f / (float)ROWS);

  // 7) head
  head_kernel<<<ROWS / 8, 256, 0, stream>>>(Y2, scl2, shf2, W3, b3, Q_out);

  (void)n_in; (void)out_size; (void)ws_size;
}
